// Block_88141318849111
// MI455X (gfx1250) — compile-verified, matrix operand order checked by reading, not run on hardware
//
#include <hip/hip_runtime.h>
#include <hip/hip_bf16.h>
#include <math.h>

typedef __attribute__((ext_vector_type(16))) __bf16 v16bf;
typedef __attribute__((ext_vector_type(8)))  float  v8f;

#define DMODEL 1024
#define NHEAD  16
#define DHEAD  64
#define FFDIM  4096
#define BATCH  2
#define SLEN   2048
#define MROWS  (BATCH * SLEN)   // 4096

union Frag16 { v16bf v; unsigned u[8]; unsigned short h[16]; };
union FragF  { v8f  v; float f[8]; };

static __device__ inline unsigned short f2bfu(float f) {
    unsigned u = __float_as_uint(f);
    // round-to-nearest-even bf16
    unsigned r = u + 0x7FFFu + ((u >> 16) & 1u);
    return (unsigned short)(r >> 16);
}

static __device__ inline v8f wmma_bf16(const v16bf a, const v16bf b, const v8f c) {
    return __builtin_amdgcn_wmma_f32_16x16x32_bf16(false, a, false, b, (short)0, c, false, false);
}

// lane-wise async 16-byte copy global -> LDS (ASYNCcnt-tracked, no VGPR data)
static __device__ inline void async_copy_b128(const void* gptr, const void* lptr) {
    unsigned lds = (unsigned)(size_t)lptr;           // flat LDS addr[31:0] = LDS offset
    unsigned long long ga = (unsigned long long)(size_t)gptr;
    asm volatile("global_load_async_to_lds_b128 %0, %1, off"
                 :: "v"(lds), "v"(ga) : "memory");
}

// ---------------------------------------------------------------------------
// fp32 -> bf16 elementwise conversion (for weights)
// ---------------------------------------------------------------------------
__global__ void cvt_f32_bf16_kernel(const float* __restrict__ src,
                                    unsigned short* __restrict__ dst, int n) {
    int i = blockIdx.x * blockDim.x + threadIdx.x;
    if (i < n) dst[i] = f2bfu(src[i]);
}

// ---------------------------------------------------------------------------
// LayerNorm over D=1024, one row per block, 256 threads, bf16 output
// ---------------------------------------------------------------------------
__global__ __launch_bounds__(256)
void layernorm_bf16_kernel(const float* __restrict__ x,
                           const float* __restrict__ g,
                           const float* __restrict__ bb,
                           unsigned short* __restrict__ out) {
    __shared__ float red[256];
    const int row = blockIdx.x;
    const float* xr = x + (size_t)row * DMODEL;
    float vals[4];
    float s = 0.f;
#pragma unroll
    for (int i = 0; i < 4; ++i) {
        vals[i] = xr[threadIdx.x + 256 * i];
        s += vals[i];
    }
    red[threadIdx.x] = s;
    __syncthreads();
    for (int off = 128; off > 0; off >>= 1) {
        if (threadIdx.x < off) red[threadIdx.x] += red[threadIdx.x + off];
        __syncthreads();
    }
    float mu = red[0] * (1.0f / DMODEL);
    __syncthreads();
    float vs = 0.f;
#pragma unroll
    for (int i = 0; i < 4; ++i) { float d = vals[i] - mu; vs += d * d; }
    red[threadIdx.x] = vs;
    __syncthreads();
    for (int off = 128; off > 0; off >>= 1) {
        if (threadIdx.x < off) red[threadIdx.x] += red[threadIdx.x + off];
        __syncthreads();
    }
    float rstd = rsqrtf(red[0] * (1.0f / DMODEL) + 1e-5f);
#pragma unroll
    for (int i = 0; i < 4; ++i) {
        int c = threadIdx.x + 256 * i;
        out[(size_t)row * DMODEL + c] = f2bfu((vals[i] - mu) * rstd * g[c] + bb[c]);
    }
}

// ---------------------------------------------------------------------------
// bf16 GEMM: C[M,N] = A[M,K] @ B[K,N] + bias (+resid) (gelu?)
// 128x64 block tile, 8 waves in 4(M)x2(N) grid; each wave owns a 32x32
// subtile = four 16x16 WMMA accumulators (A/B fragments reused 2x each).
// A tile staged via global_load_async_to_lds_b128 (ASYNCcnt path);
// B tile staged transposed via regular loads + ds_store_b16.
// ---------------------------------------------------------------------------
__global__ __launch_bounds__(256)
void gemm_bf16_kernel(const unsigned short* __restrict__ A,
                      const unsigned short* __restrict__ Bm,
                      const float* __restrict__ bias,
                      const float* __restrict__ resid,     // nullable
                      float* __restrict__ outF,            // nullable
                      unsigned short* __restrict__ outB,   // nullable
                      int M, int N, int K, int doGelu) {
    __shared__ unsigned short shA[128][40];  // row stride 80B (16B aligned, odd-bank)
    __shared__ unsigned short shB[64][34];   // [n][k] transposed, even stride

    const int bn   = blockIdx.x * 64;
    const int bm   = blockIdx.y * 128;
    const int wave = threadIdx.x >> 5;
    const int lane = threadIdx.x & 31;
    const int half = lane >> 4;
    const int l16  = lane & 15;
    const int wm   = wave & 3;   // M slab (32 rows)
    const int wn   = wave >> 2;  // N slab (32 cols)

    FragF acc[2][2];
#pragma unroll
    for (int ms = 0; ms < 2; ++ms)
#pragma unroll
        for (int ns = 0; ns < 2; ++ns) {
            v8f z = {0.f, 0.f, 0.f, 0.f, 0.f, 0.f, 0.f, 0.f};
            acc[ms][ns].v = z;
        }

    for (int k0 = 0; k0 < K; k0 += 32) {
        const int tid = threadIdx.x;
        // A tile: 128 rows x 32 bf16 (64B/row) = 512 x 16B chunks, async copy
#pragma unroll
        for (int i = 0; i < 2; ++i) {
            int idx = tid * 2 + i;          // 0..511
            int r = idx >> 2, c = idx & 3;  // row, 16B chunk
            async_copy_b128(A + (size_t)(bm + r) * K + k0 + c * 8, &shA[r][c * 8]);
        }
        // B tile: 32 k-rows x 64 n-cols, stored transposed into shB[n][k]
#pragma unroll
        for (int i = 0; i < 4; ++i) {
            int idx = tid * 4 + i;          // 0..1023
            int r = idx >> 5, c = idx & 31;
            unsigned vv = *(const unsigned*)(Bm + (size_t)(k0 + r) * N + bn + 2 * c);
            shB[2 * c][r]     = (unsigned short)(vv & 0xFFFFu);
            shB[2 * c + 1][r] = (unsigned short)(vv >> 16);
        }
        asm volatile("s_wait_asynccnt 0" ::: "memory");
        __syncthreads();

        Frag16 af[2], bf[2];
#pragma unroll
        for (int ms = 0; ms < 2; ++ms)
#pragma unroll
            for (int v = 0; v < 8; ++v) {
                int k = ((v < 4) ? 0 : 16) + half * 8 + 2 * (v & 3);
                af[ms].u[v] = *(const unsigned*)&shA[wm * 32 + ms * 16 + l16][k];
            }
#pragma unroll
        for (int ns = 0; ns < 2; ++ns) {
            int n = wn * 32 + ns * 16 + l16;
#pragma unroll
            for (int v = 0; v < 8; ++v) {
                int k = ((v < 4) ? 0 : 16) + half * 8 + 2 * (v & 3);
                bf[ns].u[v] = *(const unsigned*)&shB[n][k];
            }
        }
#pragma unroll
        for (int ms = 0; ms < 2; ++ms)
#pragma unroll
            for (int ns = 0; ns < 2; ++ns)
                acc[ms][ns].v = wmma_bf16(af[ms].v, bf[ns].v, acc[ms][ns].v);
        __syncthreads();
    }

    // epilogue: C/D layout -> lane l16 = col-in-tile, row = v + 8*half
#pragma unroll
    for (int ms = 0; ms < 2; ++ms)
#pragma unroll
        for (int ns = 0; ns < 2; ++ns) {
            int col = bn + wn * 32 + ns * 16 + l16;
            float bcol = bias[col];
#pragma unroll
            for (int v = 0; v < 8; ++v) {
                int row = bm + wm * 32 + ms * 16 + v + 8 * half;
                float val = acc[ms][ns].f[v] + bcol;
                if (doGelu) {
                    float x3 = val * val * val;
                    val = 0.5f * val *
                          (1.f + tanhf(0.7978845608028654f * (val + 0.044715f * x3)));
                }
                size_t o = (size_t)row * N + col;
                if (resid) val += resid[o];
                if (outF) outF[o] = val;
                if (outB) outB[o] = f2bfu(val);
            }
        }
}

// ---------------------------------------------------------------------------
// Flash attention. qkv is bf16 [B,S,3D] (q|k|v per row). 8 waves/block,
// each wave owns 16 query rows; block covers 128 q rows for one (b,h).
// Key loop in steps of 32 keys. QK^T and PV both on v_wmma bf16.
// ---------------------------------------------------------------------------
__global__ __launch_bounds__(256)
void attn_kernel(const unsigned short* __restrict__ qkv,
                 unsigned short* __restrict__ attnOut) {
    __shared__ unsigned       shK[32][33];      // [key][dh-pair]
    __shared__ unsigned short shV[64][34];      // [dh][key]  (transposed)
    __shared__ unsigned short shP[8][16][32];   // per-wave P transpose buffer

    const int bh   = blockIdx.y;        // 0..B*H-1
    const int b    = bh >> 4;
    const int h    = bh & 15;
    const int q0   = blockIdx.x * 128;
    const int wave = threadIdx.x >> 5;
    const int lane = threadIdx.x & 31;
    const int half = lane >> 4;
    const int l16  = lane & 15;

    const size_t rowStride = 3 * DMODEL;
    const size_t baseQ = (size_t)b * SLEN * rowStride + h * DHEAD;
    const size_t baseK = baseQ + DMODEL;
    const size_t baseV = baseQ + 2 * DMODEL;

    // Q fragments: rows m = l16, two K32 chunks covering Dh=64
    Frag16 qf[2];
    {
        const unsigned short* qrow = qkv + baseQ + (size_t)(q0 + wave * 16 + l16) * rowStride;
#pragma unroll
        for (int v = 0; v < 8; ++v) {
            int k = ((v < 4) ? 0 : 16) + half * 8 + 2 * (v & 3);
            qf[0].u[v] = *(const unsigned*)(qrow + k);
            qf[1].u[v] = *(const unsigned*)(qrow + 32 + k);
        }
    }

    FragF of[4];
#pragma unroll
    for (int t = 0; t < 4; ++t) {
        v8f z = {0.f, 0.f, 0.f, 0.f, 0.f, 0.f, 0.f, 0.f};
        of[t].v = z;
    }
    float mrow[8], lrow[8];
#pragma unroll
    for (int v = 0; v < 8; ++v) { mrow[v] = -1e30f; lrow[v] = 0.f; }

    const float scale = 0.125f;  // 1/sqrt(64)

    for (int kt = 0; kt < SLEN / 32; ++kt) {
        // cooperative stage: 32 keys x 64 dh of K and V (V transposed)
        const int tid = threadIdx.x;
#pragma unroll
        for (int i = 0; i < 4; ++i) {
            int idx = tid * 4 + i;          // 0..1023
            int r = idx >> 5;               // key 0..31
            int c = idx & 31;               // dh pair 0..31
            int srow = kt * 32 + r;
            shK[r][c] = *(const unsigned*)(qkv + baseK + (size_t)srow * rowStride + 2 * c);
            unsigned vv = *(const unsigned*)(qkv + baseV + (size_t)srow * rowStride + 2 * c);
            shV[2 * c][r]     = (unsigned short)(vv & 0xFFFFu);
            shV[2 * c + 1][r] = (unsigned short)(vv >> 16);
        }
        __syncthreads();

        // scores: two 16-key tiles, contraction over dh (2 x K32)
        FragF sf[2];
#pragma unroll
        for (int t = 0; t < 2; ++t) {
            Frag16 kb0, kb1;
            int key = t * 16 + l16;
#pragma unroll
            for (int v = 0; v < 8; ++v) {
                int k = ((v < 4) ? 0 : 16) + half * 8 + 2 * (v & 3);
                kb0.u[v] = shK[key][k >> 1];
                kb1.u[v] = shK[key][(k >> 1) + 16];
            }
            v8f z = {0.f, 0.f, 0.f, 0.f, 0.f, 0.f, 0.f, 0.f};
            v8f a = wmma_bf16(qf[0].v, kb0.v, z);
            sf[t].v = wmma_bf16(qf[1].v, kb1.v, a);
        }

        // online softmax; rows of D-layout are (v + 8*half), cols spread
        // across the 16 lanes of each half-wave -> xor-shuffle reductions
        float mnew[8], alpha[8];
#pragma unroll
        for (int v = 0; v < 8; ++v) {
            float s0 = sf[0].f[v] * scale;
            float s1 = sf[1].f[v] * scale;
            sf[0].f[v] = s0; sf[1].f[v] = s1;
            float mx = fmaxf(s0, s1);
            for (int off = 1; off < 16; off <<= 1)
                mx = fmaxf(mx, __shfl_xor(mx, off, 32));
            mnew[v]  = fmaxf(mrow[v], mx);
            alpha[v] = __expf(mrow[v] - mnew[v]);
            mrow[v]  = mnew[v];
        }
#pragma unroll
        for (int v = 0; v < 8; ++v) {
            float p0 = __expf(sf[0].f[v] - mnew[v]);
            float p1 = __expf(sf[1].f[v] - mnew[v]);
            float sum = p0 + p1;
            for (int off = 1; off < 16; off <<= 1)
                sum += __shfl_xor(sum, off, 32);
            lrow[v] = lrow[v] * alpha[v] + sum;
            int m = v + 8 * half;
            shP[wave][m][l16]      = f2bfu(p0);
            shP[wave][m][16 + l16] = f2bfu(p1);
#pragma unroll
            for (int t = 0; t < 4; ++t) of[t].f[v] *= alpha[v];
        }
        asm volatile("s_wait_dscnt 0" ::: "memory");

        // P in A-fragment layout (contraction over 32 keys)
        Frag16 pf;
#pragma unroll
        for (int v = 0; v < 8; ++v) {
            int k = ((v < 4) ? 0 : 16) + half * 8 + 2 * (v & 3);
            pf.u[v] = *(const unsigned*)&shP[wave][l16][k];
        }
        // accumulate P @ V over four 16-wide dh tiles
#pragma unroll
        for (int t = 0; t < 4; ++t) {
            Frag16 vf;
#pragma unroll
            for (int v = 0; v < 8; ++v) {
                int k = ((v < 4) ? 0 : 16) + half * 8 + 2 * (v & 3);
                vf.u[v] = *(const unsigned*)&shV[t * 16 + l16][k];
            }
            of[t].v = wmma_bf16(pf.v, vf.v, of[t].v);
        }
        __syncthreads();
    }

    // finalize: divide by row sums, write bf16 [B,S,D] with heads interleaved
    float inv[8];
#pragma unroll
    for (int v = 0; v < 8; ++v) inv[v] = 1.0f / lrow[v];
    const int srowBase = q0 + wave * 16;
#pragma unroll
    for (int t = 0; t < 4; ++t) {
        int dh = t * 16 + l16;
#pragma unroll
        for (int v = 0; v < 8; ++v) {
            int srow = srowBase + v + 8 * half;
            attnOut[((size_t)b * SLEN + srow) * DMODEL + h * DHEAD + dh] =
                f2bfu(of[t].f[v] * inv[v]);
        }
    }
}

// ---------------------------------------------------------------------------
// Host-side orchestration
// ---------------------------------------------------------------------------
extern "C" void kernel_launch(void* const* d_in, const int* in_sizes, int n_in,
                              void* d_out, int out_size, void* d_ws, size_t ws_size,
                              hipStream_t stream) {
    (void)in_sizes; (void)n_in; (void)out_size; (void)ws_size;

    const float* x     = (const float*)d_in[0];
    const float* ln1_g = (const float*)d_in[1];
    const float* ln1_b = (const float*)d_in[2];
    const float* Wqkv  = (const float*)d_in[3];
    const float* bqkv  = (const float*)d_in[4];
    const float* Wo    = (const float*)d_in[5];
    const float* bo    = (const float*)d_in[6];
    const float* ln2_g = (const float*)d_in[7];
    const float* ln2_b = (const float*)d_in[8];
    const float* W1    = (const float*)d_in[9];
    const float* b1    = (const float*)d_in[10];
    const float* W2    = (const float*)d_in[11];
    const float* b2    = (const float*)d_in[12];
    float* out = (float*)d_out;

    char* ws = (char*)d_ws;
    size_t off = 0;
    auto take = [&](size_t bytes) -> void* {
        void* p = ws + off;
        off = (off + bytes + 255) & ~(size_t)255;
        return p;
    };

    unsigned short* WqkvB = (unsigned short*)take((size_t)DMODEL * 3 * DMODEL * 2);
    unsigned short* WoB   = (unsigned short*)take((size_t)DMODEL * DMODEL * 2);
    unsigned short* W1B   = (unsigned short*)take((size_t)DMODEL * FFDIM * 2);
    unsigned short* W2B   = (unsigned short*)take((size_t)FFDIM * DMODEL * 2);
    unsigned short* h1B   = (unsigned short*)take((size_t)MROWS * DMODEL * 2);
    unsigned short* qkvB  = (unsigned short*)take((size_t)MROWS * 3 * DMODEL * 2);
    unsigned short* attnB = (unsigned short*)take((size_t)MROWS * DMODEL * 2);
    float*          x1F   = (float*)take((size_t)MROWS * DMODEL * 4);
    unsigned short* h2B   = (unsigned short*)take((size_t)MROWS * DMODEL * 2);
    unsigned short* f1B   = (unsigned short*)take((size_t)MROWS * FFDIM * 2);

    auto cvt = [&](const float* s, unsigned short* d, int n) {
        cvt_f32_bf16_kernel<<<(n + 255) / 256, 256, 0, stream>>>(s, d, n);
    };
    cvt(Wqkv, WqkvB, DMODEL * 3 * DMODEL);
    cvt(Wo,   WoB,   DMODEL * DMODEL);
    cvt(W1,   W1B,   DMODEL * FFDIM);
    cvt(W2,   W2B,   FFDIM * DMODEL);

    // LN1
    layernorm_bf16_kernel<<<MROWS, 256, 0, stream>>>(x, ln1_g, ln1_b, h1B);

    // QKV projection: [4096,1024] @ [1024,3072] -> bf16 qkv
    gemm_bf16_kernel<<<dim3(3 * DMODEL / 64, MROWS / 128), 256, 0, stream>>>(
        h1B, WqkvB, bqkv, nullptr, nullptr, qkvB, MROWS, 3 * DMODEL, DMODEL, 0);

    // Flash attention -> bf16 attn
    attn_kernel<<<dim3(SLEN / 128, BATCH * NHEAD), 256, 0, stream>>>(qkvB, attnB);

    // Output projection + residual(x) -> fp32 x1
    gemm_bf16_kernel<<<dim3(DMODEL / 64, MROWS / 128), 256, 0, stream>>>(
        attnB, WoB, bo, x, x1F, nullptr, MROWS, DMODEL, DMODEL, 0);

    // LN2
    layernorm_bf16_kernel<<<MROWS, 256, 0, stream>>>(x1F, ln2_g, ln2_b, h2B);

    // FFN1 + GELU -> bf16
    gemm_bf16_kernel<<<dim3(FFDIM / 64, MROWS / 128), 256, 0, stream>>>(
        h2B, W1B, b1, nullptr, nullptr, f1B, MROWS, FFDIM, DMODEL, 1);

    // FFN2 + residual(x1) -> fp32 out
    gemm_bf16_kernel<<<dim3(DMODEL / 64, MROWS / 128), 256, 0, stream>>>(
        f1B, W2B, b2, x1F, out, nullptr, MROWS, DMODEL, FFDIM, 0);
}
